// SequenceAttention_84731114816302
// MI455X (gfx1250) — matrix operand order repaired and checked by reading, not compiled, not run on hardware
//
#include <hip/hip_runtime.h>
#include <hip/hip_bf16.h>

typedef __attribute__((ext_vector_type(16))) __bf16 v16bf;
typedef __attribute__((ext_vector_type(8)))  __bf16 v8bf;
typedef __attribute__((ext_vector_type(8)))  float  v8f;

#define BATCH 16
#define SEQ   2048
#define DIM   512
#define TAU   0.1f
#define NEGV  -1000000.0f
#define SCALE 0.044194173824159216f   // 1/sqrt(512)
#define LDSPITCH (SEQ + 17)           // padded LDS row stride (floats)

// ---------------------------------------------------------------------------
// Fragment loaders. k order used here for WMMA 16x16x32 bf16 operands (same for A and B):
//   A (16x32, MxK): lane m = lane&15, kh = lane>>4;
//     elems 0..15 -> K = kbase + kh*16 + e   (contiguous)
//   B (32x16, KxN): lane n = lane&15, kh = lane>>4;
//     elems 0..15 -> K = kbase + kh*16 + e   (contiguous)
// ---------------------------------------------------------------------------
__device__ __forceinline__ v16bf load_a_frag_bf16(const __bf16* __restrict__ base,
                                                  int ld, int m, int kh, int kbase) {
  const __bf16* p = base + (size_t)m * ld + kbase + kh * 16;
  v8bf lo = *(const v8bf*)p;          // 16B
  v8bf hi = *(const v8bf*)(p + 8);    // 16B
  v16bf f;
#pragma unroll
  for (int i = 0; i < 8; ++i) { f[i] = lo[i]; f[8 + i] = hi[i]; }
  return f;
}

__device__ __forceinline__ v16bf load_b_frag_bf16(const __bf16* __restrict__ base,
                                                  int ld, int n, int kh, int kbase) {
  return *(const v16bf*)(base + (size_t)n * ld + kbase + kh * 16);  // 32B
}

// A fragment from f32 source with inline bf16 conversion (kernel 2, weights).
__device__ __forceinline__ v16bf load_a_frag_f32(const float* __restrict__ base,
                                                 int ld, int m, int kh, int kbase) {
  const float* p0 = base + (size_t)m * ld + kbase + kh * 16;
  const float* p1 = p0 + 8;
  v16bf f;
#pragma unroll
  for (int i = 0; i < 8; ++i) f[i] = (__bf16)p0[i];
#pragma unroll
  for (int i = 0; i < 8; ++i) f[8 + i] = (__bf16)p1[i];
  return f;
}

// ---------------------------------------------------------------------------
// Kernel 0: one pass over x produces BOTH bf16 copies:
//   xb[b][s][d] (row-major, for kernel 1 A/B)  and
//   xt[b][d][s] (transposed, for kernel 2 B), via 32x32 LDS tile.
// ---------------------------------------------------------------------------
__global__ __launch_bounds__(256) void cvt_dual_bf16(
    const float* __restrict__ x, __bf16* __restrict__ xb, __bf16* __restrict__ xt) {
  __shared__ __bf16 tile[32][33];
  const int b  = blockIdx.z;
  const int s0 = blockIdx.x * 32;
  const int d0 = blockIdx.y * 32;
  const int tx = threadIdx.x & 31;
  const int ty = threadIdx.x >> 5;  // 0..7

  const float* xp = x  + ((size_t)b * SEQ + s0) * DIM + d0;
  __bf16*      rp = xb + ((size_t)b * SEQ + s0) * DIM + d0;
#pragma unroll
  for (int i = ty; i < 32; i += 8) {
    __bf16 v = (__bf16)xp[(size_t)i * DIM + tx];
    tile[i][tx] = v;                    // [s_local][d_local]
    rp[(size_t)i * DIM + tx] = v;       // row-major bf16 copy (coalesced)
  }
  __syncthreads();
  __bf16* op = xt + ((size_t)b * DIM + d0) * SEQ + s0;
#pragma unroll
  for (int i = ty; i < 32; i += 8)
    op[(size_t)i * SEQ + tx] = tile[tx][i];  // row d0+i, col s0+tx
}

// ---------------------------------------------------------------------------
// Kernel 1: scores = (Xblk . X^T)/sqrt(D), mask (>TAU else NEG), softmax,
//           write weights. One workgroup per (batch, 16-row block).
// 8 waves; wave w computes column tiles jt = w, w+8, ... (128 tiles of 16).
// All GEMM operands are pre-converted bf16 (pure b128 loads in hot loop).
// Full 16x2048 masked-score row block staged in LDS (132 KB < 320 KB/WGP).
// ---------------------------------------------------------------------------
__global__ __launch_bounds__(256) void attn_scores_softmax(
    const __bf16* __restrict__ xb, float* __restrict__ wts) {
  __shared__ float sc[16 * LDSPITCH];

  const int b       = blockIdx.y;
  const int rowbase = blockIdx.x * 16;
  const int tid     = threadIdx.x;
  const int wave    = tid >> 5;
  const int lane    = tid & 31;
  const int hn      = lane & 15;   // row (A) / col (B,C) within tile
  const int kh      = lane >> 4;

  const __bf16* xbb = xb + (size_t)b * SEQ * DIM;

  // Hoist A fragments for the whole K=512 of this 16-row block (16 x v16bf).
  v16bf afrag[16];
#pragma unroll
  for (int t = 0; t < 16; ++t)
    afrag[t] = load_a_frag_bf16(xbb + (size_t)rowbase * DIM, DIM, hn, kh, t * 32);

  for (int jt = wave; jt < SEQ / 16; jt += 8) {
    const int colbase = jt * 16;
    const __bf16* brow = xbb + (size_t)colbase * DIM;
    v8f acc = {};
#pragma unroll
    for (int t = 0; t < 16; ++t) {
      v16bf bfrag = load_b_frag_bf16(brow, DIM, hn, kh, t * 32);
      acc = __builtin_amdgcn_wmma_f32_16x16x32_bf16(
          false, afrag[t], false, bfrag, (short)0, acc, false, false);
    }
    // C layout: VGPR r -> row M = r + kh*8, col N = hn. Scale + mask -> LDS.
#pragma unroll
    for (int r = 0; r < 8; ++r) {
      float s = acc[r] * SCALE;
      s = (s > TAU) ? s : NEGV;
      sc[(kh * 8 + r) * LDSPITCH + colbase + hn] = s;
    }
  }
  __syncthreads();

  // Softmax: row r handled by 16 threads (tid>>4 == r), lanes tid&15.
  const int r  = tid >> 4;
  const int c0 = tid & 15;
  float* srow = &sc[r * LDSPITCH];

  float mx = -3.4e38f;
  for (int c = c0; c < SEQ; c += 16) mx = fmaxf(mx, srow[c]);
#pragma unroll
  for (int off = 8; off; off >>= 1) mx = fmaxf(mx, __shfl_xor(mx, off, 16));

  float sum = 0.f;
  for (int c = c0; c < SEQ; c += 16) {
    float e = __expf(srow[c] - mx);
    srow[c] = e;
    sum += e;
  }
#pragma unroll
  for (int off = 8; off; off >>= 1) sum += __shfl_xor(sum, off, 16);

  const float inv = 1.0f / sum;
  float* wrow = wts + ((size_t)b * SEQ + rowbase + r) * SEQ;
  for (int c = c0; c < SEQ; c += 16) wrow[c] = srow[c] * inv;
}

// ---------------------------------------------------------------------------
// Kernel 2: context = weights . X   (M=2048 rows, N=512 cols, K=2048)
// One workgroup per (batch, 16-row block); wave w owns 4 column tiles
// (w + 8p)*16, sharing one A fragment (weights f32 -> bf16 inline) per K-step.
// B operand from pre-transposed bf16 xt[b][d][t] (contiguous in K).
// ---------------------------------------------------------------------------
__global__ __launch_bounds__(256) void attn_context(
    const float* __restrict__ wts, const __bf16* __restrict__ xt,
    float* __restrict__ ctx) {
  const int b       = blockIdx.y;
  const int rowbase = blockIdx.x * 16;
  const int tid     = threadIdx.x;
  const int wave    = tid >> 5;
  const int lane    = tid & 31;
  const int hn      = lane & 15;
  const int kh      = lane >> 4;

  const float*  wb  = wts + ((size_t)b * SEQ + rowbase) * SEQ;  // A, ld=SEQ
  const __bf16* xtb = xt + (size_t)b * DIM * SEQ;               // [d][t]

  v8f acc[4] = {};
  for (int t = 0; t < SEQ / 32; ++t) {
    v16bf af = load_a_frag_f32(wb, SEQ, hn, kh, t * 32);
    __builtin_prefetch(wb + (size_t)hn * SEQ + t * 32 + 256, 0, 1);
#pragma unroll
    for (int p = 0; p < 4; ++p) {
      const int colbase = (wave + 8 * p) * 16;
      const __bf16* bp = xtb + (size_t)(colbase + hn) * SEQ + t * 32 + kh * 16;
      v16bf bf = *(const v16bf*)bp;
      acc[p] = __builtin_amdgcn_wmma_f32_16x16x32_bf16(
          false, af, false, bf, (short)0, acc[p], false, false);
    }
  }

#pragma unroll
  for (int p = 0; p < 4; ++p) {
    const int colbase = (wave + 8 * p) * 16;
    float* cp = ctx + ((size_t)b * SEQ + rowbase) * DIM + colbase;
#pragma unroll
    for (int r = 0; r < 8; ++r)
      cp[(size_t)(kh * 8 + r) * DIM + hn] = acc[p][r];
  }
}

extern "C" void kernel_launch(void* const* d_in, const int* in_sizes, int n_in,
                              void* d_out, int out_size, void* d_ws, size_t ws_size,
                              hipStream_t stream) {
  const float* x = (const float*)d_in[0];
  float* ctx = (float*)d_out;                                   // [16,2048,512]
  float* wts = (float*)d_out + (size_t)BATCH * SEQ * DIM;       // [16,2048,2048]
  __bf16* xb = (__bf16*)d_ws;                                   // 33.5 MB row-major bf16
  __bf16* xt = xb + (size_t)BATCH * SEQ * DIM;                  // 33.5 MB transposed bf16

  cvt_dual_bf16<<<dim3(SEQ / 32, DIM / 32, BATCH), 256, 0, stream>>>(x, xb, xt);
  attn_scores_softmax<<<dim3(SEQ / 16, BATCH), 256, 0, stream>>>(xb, wts);
  attn_context<<<dim3(SEQ / 16, BATCH), 256, 0, stream>>>(wts, xt, ctx);
}
